// InstLM_50225347559519
// MI455X (gfx1250) — 16-bit and 8-bit matrix operand order repaired and checked by reading, compiled, not run on hardware
//
#include <hip/hip_runtime.h>
#include <hip/hip_bf16.h>
#include <math.h>

typedef __attribute__((ext_vector_type(16))) _Float16 v16h;
typedef __attribute__((ext_vector_type(8)))  _Float16 v8h;
typedef __attribute__((ext_vector_type(8)))  float    v8f;
typedef __attribute__((ext_vector_type(16))) int      v16i;
typedef __attribute__((ext_vector_type(4)))  int      v4i;

#define DEV __device__ __forceinline__

DEV float sigf(float x) {
  x = fminf(fmaxf(x, -30.0f), 30.0f);
  return __builtin_amdgcn_rcpf(1.0f + __expf(-x));
}
DEV float tanh_fast(float x) {
  x = fminf(fmaxf(x, -15.0f), 15.0f);
  float e = __expf(2.0f * x);
  return (e - 1.0f) * __builtin_amdgcn_rcpf(e + 1.0f);
}

DEV unsigned char f32_to_e4m3(float x) {
  float ax = fabsf(x);
  unsigned s = (__float_as_uint(x) & 0x80000000u) >> 24;
  if (ax >= 448.0f) return (unsigned char)(s | 0x7E);
  if (ax < 0.015625f) {
    int m = (int)(ax * 512.0f + 0.5f);
    return (unsigned char)(s | m);
  }
  unsigned b = __float_as_uint(ax);
  int e = (int)((b >> 23) & 0xFF) - 127;
  unsigned m3 = ((b & 0x7FFFFFu) + 0x80000u) >> 20;
  int E = e + 7;
  if (m3 == 8) { m3 = 0; ++E; }
  return (unsigned char)(s | (E << 3) | m3);
}

DEV v16h load_A_lds(const _Float16* As, int lda, int k0, int lane) {
  int row = lane & 15, g = lane >> 4;
  const _Float16* p = As + row * lda + k0 + (g << 4);
  v8h lo = *(const v8h*)(p);
  v8h hi = *(const v8h*)(p + 8);
  v16h a;
#pragma unroll
  for (int j = 0; j < 8; ++j) { a[j] = lo[j]; a[j + 8] = hi[j]; }
  return a;
}

DEV v16h load_B_glb(const _Float16* W, int ldk, int n0, int k0, int lane) {
  int n = lane & 15, g = lane >> 4;
  return *(const v16h*)(W + (size_t)(n0 + n) * ldk + k0 + (g << 4));
}

DEV v16i load_A8_lds(const unsigned char* As8, int lane) {
  int row = lane & 15, g = lane >> 4;
  const unsigned char* p = As8 + row * 128 + (g << 4);
  v16i a;
#pragma unroll
  for (int pr = 0; pr < 8; ++pr) {
    const int* q = (const int*)(p + (pr >> 1) * 32 + (pr & 1) * 8);
    a[2 * pr] = q[0];
    a[2 * pr + 1] = q[1];
  }
  return a;
}

DEV v16i load_B8_glb(const unsigned char* W, int n0, int lane) {
  int n = lane & 15, g = lane >> 4;
  const unsigned char* p = W + (size_t)(n0 + n) * 128 + (g << 4);
  v16i b;
#pragma unroll
  for (int q = 0; q < 4; ++q) {
    v4i c = *(const v4i*)(p + q * 32);
    b[4 * q + 0] = c[0]; b[4 * q + 1] = c[1];
    b[4 * q + 2] = c[2]; b[4 * q + 3] = c[3];
  }
  return b;
}

DEV v8f wmma16(v16h a, v16h b, v8f c) {
  return __builtin_amdgcn_wmma_f32_16x16x32_f16(false, a, false, b, (short)0, c,
                                                false, false);
}
DEV v8f wmma8(v16i a, v16i b, v8f c) {
  return __builtin_amdgcn_wmma_f32_16x16x128_fp8_fp8(a, b, (short)0, c, false,
                                                     false);
}

__global__ __launch_bounds__(256) void prep_wout_kernel(
    const float* __restrict__ Wout, const float* __restrict__ bout,
    _Float16* __restrict__ W16, unsigned char* __restrict__ W8,
    float* __restrict__ bpad) {
  int idx = blockIdx.x * blockDim.x + threadIdx.x;
  if (idx < 10016 * 128) {
    int n = idx >> 7;
    float v = (n < 10001) ? Wout[idx] : 0.0f;
    W16[idx] = (_Float16)v;
    W8[idx] = f32_to_e4m3(v);
  }
  if (idx < 10016) bpad[idx] = (idx < 10001) ? bout[idx] : -1.0e30f;
}

__global__ __launch_bounds__(256) void prep_small_kernel(
    const float* __restrict__ Wih, const float* __restrict__ Whh,
    const float* __restrict__ bih, const float* __restrict__ bhh,
    const float* __restrict__ ceWih, const float* __restrict__ ceWhh,
    const float* __restrict__ cebih, const float* __restrict__ cebhh,
    _Float16* __restrict__ Wmain, float* __restrict__ bmain,
    _Float16* __restrict__ Wce, float* __restrict__ bce) {
  int idx = blockIdx.x * blockDim.x + threadIdx.x;
  if (idx < 512 * 192) {
    int n = idx / 192, k = idx % 192;
    Wmain[idx] = (_Float16)((k < 64) ? Wih[n * 64 + k] : Whh[n * 128 + k - 64]);
  }
  if (idx < 256 * 128) {
    int n = idx >> 7, k = idx & 127;
    Wce[idx] = (_Float16)((k < 64) ? ceWih[n * 64 + k] : ceWhh[n * 64 + k - 64]);
  }
  if (idx < 512) bmain[idx] = bih[idx] + bhh[idx];
  if (idx < 256) bce[idx] = cebih[idx] + cebhh[idx];
}

__global__ __launch_bounds__(256) void canvas_encode_kernel(
    const float* __restrict__ prevc, const float* __restrict__ finalc,
    const float* __restrict__ Wobj, const float* __restrict__ bobj,
    const _Float16* __restrict__ Wce, const float* __restrict__ bce,
    float* __restrict__ enc_out) {
  __shared__ __align__(32) _Float16 As[16][128];
  __shared__ float gates[16][256];
  __shared__ float hbuf[16][64], cbuf[16][64], encacc[16][64];
  __shared__ float seq[16][25][4];
  int tid = threadIdx.x, lane = tid & 31, wave = tid >> 5;
  int bb = blockIdx.x << 4;

  v16h Bce[2][4];
#pragma unroll
  for (int it = 0; it < 2; ++it)
#pragma unroll
    for (int kk = 0; kk < 4; ++kk)
      Bce[it][kk] = load_B_glb(Wce, 128, ((wave << 1) + it) << 4, kk << 5, lane);

  for (int r = tid; r < 16 * 64; r += 256) (&encacc[0][0])[r] = 0.0f;

  for (int cidx = 0; cidx < 2; ++cidx) {
    const float* canvas = cidx ? finalc : prevc;
    __syncthreads();
    if (tid < 16) {
      int b = bb + tid, cnt = 0;
      for (int o = 0; o < 25; ++o) {
        float v0 = canvas[((size_t)b * 25 + o) * 4 + 0];
        float v1 = canvas[((size_t)b * 25 + o) * 4 + 1];
        float v2 = canvas[((size_t)b * 25 + o) * 4 + 2];
        float v3 = canvas[((size_t)b * 25 + o) * 4 + 3];
        if (v0 + v1 + v2 + v3 >= 0.0f) {
          seq[tid][cnt][0] = v0; seq[tid][cnt][1] = v1;
          seq[tid][cnt][2] = v2; seq[tid][cnt][3] = v3; ++cnt;
        }
      }
      for (; cnt < 25; ++cnt)
        for (int d = 0; d < 4; ++d) seq[tid][cnt][d] = -1.0f;
    }
    for (int r = tid; r < 16 * 64; r += 256) {
      (&hbuf[0][0])[r] = 0.0f; (&cbuf[0][0])[r] = 0.0f;
    }
    __syncthreads();
    for (int r = tid; r < 16 * 64; r += 256) {
      int m = r >> 6, j = r & 63;
      float x = bobj[j];
      for (int d = 0; d < 4; ++d) x += seq[m][0][d] * Wobj[j * 4 + d];
      As[m][j] = (_Float16)x;
      As[m][64 + j] = (_Float16)0.0f;
    }
    __syncthreads();

    for (int s = 0; s < 25; ++s) {
      v16h Af[4];
#pragma unroll
      for (int kk = 0; kk < 4; ++kk)
        Af[kk] = load_A_lds(&As[0][0], 128, kk << 5, lane);
#pragma unroll
      for (int it = 0; it < 2; ++it) {
        int n0 = ((wave << 1) + it) << 4;
        v8f acc = {};
#pragma unroll
        for (int kk = 0; kk < 4; ++kk) acc = wmma16(Af[kk], Bce[it][kk], acc);
        int n = n0 + (lane & 15), g = lane >> 4;
#pragma unroll
        for (int i = 0; i < 8; ++i) gates[(g << 3) + i][n] = acc[i];
      }
      __syncthreads();
      for (int r = tid; r < 16 * 64; r += 256) {
        int m = r >> 6, j = r & 63;
        float gi = sigf(gates[m][j] + bce[j]);
        float gf = sigf(gates[m][64 + j] + bce[64 + j]);
        float gg = tanh_fast(gates[m][128 + j] + bce[128 + j]);
        float go = sigf(gates[m][192 + j] + bce[192 + j]);
        float cc = gf * cbuf[m][j] + gi * gg;
        float hh = go * tanh_fast(cc);
        cbuf[m][j] = cc; hbuf[m][j] = hh;
        As[m][64 + j] = (_Float16)hh;
        if (s < 24) {
          float x = bobj[j];
          for (int d = 0; d < 4; ++d) x += seq[m][s + 1][d] * Wobj[j * 4 + d];
          As[m][j] = (_Float16)x;
        }
      }
      __syncthreads();
    }
    for (int r = tid; r < 16 * 64; r += 256) {
      int m = r >> 6, j = r & 63;
      encacc[m][j] += hbuf[m][j];
    }
  }
  __syncthreads();
  for (int r = tid; r < 16 * 64; r += 256) {
    int m = r >> 6, j = r & 63;
    enc_out[(size_t)(bb + m) * 64 + j] = encacc[m][j];
  }
}

__global__ __launch_bounds__(256) void main_lstm_kernel(
    const int* __restrict__ inst, const float* __restrict__ tobj,
    const float* __restrict__ Wtgt, const float* __restrict__ btgt,
    const float* __restrict__ embedW, const float* __restrict__ enc,
    const _Float16* __restrict__ Wmain, const float* __restrict__ bmain,
    _Float16* __restrict__ Hseq) {
  __shared__ __align__(32) _Float16 As[16][192];
  __shared__ float gates[16][512];
  __shared__ float hbuf[16][128], cbuf[16][128];
  int tid = threadIdx.x, lane = tid & 31, wave = tid >> 5;
  int bb = blockIdx.x << 4;

  v16h Bm[4][6];
#pragma unroll
  for (int it = 0; it < 4; ++it)
#pragma unroll
    for (int kk = 0; kk < 6; ++kk)
      Bm[it][kk] = load_B_glb(Wmain, 192, ((wave << 2) + it) << 4, kk << 5, lane);

  for (int r = tid; r < 16 * 128; r += 256) {
    (&hbuf[0][0])[r] = 0.0f; (&cbuf[0][0])[r] = 0.0f;
    int m = r >> 7, j = r & 127;
    As[m][64 + j] = (_Float16)0.0f;
  }
  for (int r = tid; r < 16 * 64; r += 256) {
    int m = r >> 6, j = r & 63;
    As[m][j] = (_Float16)enc[(size_t)(bb + m) * 64 + j];
  }
  __syncthreads();

  for (int s = 0; s < 66; ++s) {
    v16h Af[6];
#pragma unroll
    for (int kk = 0; kk < 6; ++kk)
      Af[kk] = load_A_lds(&As[0][0], 192, kk << 5, lane);
#pragma unroll
    for (int it = 0; it < 4; ++it) {
      int n0 = ((wave << 2) + it) << 4;
      v8f acc = {};
#pragma unroll
      for (int kk = 0; kk < 6; ++kk) acc = wmma16(Af[kk], Bm[it][kk], acc);
      int n = n0 + (lane & 15), g = lane >> 4;
#pragma unroll
      for (int i = 0; i < 8; ++i) gates[(g << 3) + i][n] = acc[i];
    }
    __syncthreads();
    for (int r = tid; r < 16 * 128; r += 256) {
      int m = r >> 7, j = r & 127;
      float gi = sigf(gates[m][j] + bmain[j]);
      float gf = sigf(gates[m][128 + j] + bmain[128 + j]);
      float gg = tanh_fast(gates[m][256 + j] + bmain[256 + j]);
      float go = sigf(gates[m][384 + j] + bmain[384 + j]);
      float cc = gf * cbuf[m][j] + gi * gg;
      float hh = go * tanh_fast(cc);
      cbuf[m][j] = cc; hbuf[m][j] = hh;
      As[m][64 + j] = (_Float16)hh;
      if (s >= 2)
        Hseq[((size_t)(s - 2) * 256 + bb + m) * 128 + j] = (_Float16)hh;
    }
    if (s < 65) {
      for (int r = tid; r < 16 * 64; r += 256) {
        int m = r >> 6, j = r & 63;
        float x;
        if (s == 0) {
          x = btgt[j];
          for (int d = 0; d < 4; ++d)
            x += tobj[(size_t)(bb + m) * 4 + d] * Wtgt[j * 4 + d];
        } else {
          int tok = inst[(size_t)(bb + m) * 65 + (s - 1)];
          x = embedW[(size_t)tok * 64 + j];
        }
        As[m][j] = (_Float16)x;
      }
    }
    __syncthreads();
  }
}

__global__ __launch_bounds__(256) void logits_kernel(
    const _Float16* __restrict__ Hseq, const _Float16* __restrict__ Wout16,
    const unsigned char* __restrict__ Wout8, const float* __restrict__ bpad,
    float* __restrict__ out) {
  __shared__ __align__(32) _Float16 As[16][128];
  __shared__ __align__(16) unsigned char As8[16 * 128];
  __shared__ float redbuf[8][16];
  __shared__ float rowmax[16];
  __shared__ float rowlse[16];
  int tid = threadIdx.x, lane = tid & 31, wave = tid >> 5;
  int t = blockIdx.x >> 4;
  int bb = (blockIdx.x & 15) << 4;
  int n_lane = lane & 15, g = lane >> 4;

  for (int r = tid; r < 16 * 128; r += 256) {
    int m = r >> 7, j = r & 127;
    _Float16 v = Hseq[((size_t)t * 256 + bb + m) * 128 + j];
    As[m][j] = v;
    As8[m * 128 + j] = f32_to_e4m3((float)v);
  }
  __syncthreads();

  v16i A8 = load_A8_lds(As8, lane);
  v16h Af[4];
#pragma unroll
  for (int kk = 0; kk < 4; ++kk)
    Af[kk] = load_A_lds(&As[0][0], 128, kk << 5, lane);

  float rmax[8];
#pragma unroll
  for (int i = 0; i < 8; ++i) rmax[i] = -3.0e38f;
#pragma unroll 2
  for (int tile = wave; tile < 626; tile += 8) {
    int n0 = tile << 4;
    __builtin_prefetch(Wout8 + (size_t)(n0 + 256) * 128, 0, 1);
    v8f acc = {};
    acc = wmma8(A8, load_B8_glb(Wout8, n0, lane), acc);
    float bias = bpad[n0 + n_lane];
#pragma unroll
    for (int i = 0; i < 8; ++i) rmax[i] = fmaxf(rmax[i], acc[i] + bias);
  }
#pragma unroll
  for (int off = 1; off < 16; off <<= 1)
#pragma unroll
    for (int i = 0; i < 8; ++i)
      rmax[i] = fmaxf(rmax[i], __shfl_xor(rmax[i], off, 32));
  if (n_lane == 0)
#pragma unroll
    for (int i = 0; i < 8; ++i) redbuf[wave][(g << 3) + i] = rmax[i];
  __syncthreads();
  if (tid < 16) {
    float v = redbuf[0][tid];
    for (int w = 1; w < 8; ++w) v = fmaxf(v, redbuf[w][tid]);
    rowmax[tid] = v + 0.5f;
  }
  __syncthreads();

  float rsum[8], mymax[8];
#pragma unroll
  for (int i = 0; i < 8; ++i) { rsum[i] = 0.0f; mymax[i] = rowmax[(g << 3) + i]; }
#pragma unroll 2
  for (int tile = wave; tile < 626; tile += 8) {
    int n0 = tile << 4;
    v8f acc = {};
#pragma unroll
    for (int kk = 0; kk < 4; ++kk)
      acc = wmma16(Af[kk], load_B_glb(Wout16, 128, n0, kk << 5, lane), acc);
    float bias = bpad[n0 + n_lane];
#pragma unroll
    for (int i = 0; i < 8; ++i) rsum[i] += __expf(acc[i] + bias - mymax[i]);
  }
#pragma unroll
  for (int off = 1; off < 16; off <<= 1)
#pragma unroll
    for (int i = 0; i < 8; ++i) rsum[i] += __shfl_xor(rsum[i], off, 32);
  if (n_lane == 0)
#pragma unroll
    for (int i = 0; i < 8; ++i) redbuf[wave][(g << 3) + i] = rsum[i];
  __syncthreads();
  if (tid < 16) {
    float v = 0.0f;
    for (int w = 0; w < 8; ++w) v += redbuf[w][tid];
    rowlse[tid] = __logf(v);
  }
  __syncthreads();

  float myoff[8];
#pragma unroll
  for (int i = 0; i < 8; ++i)
    myoff[i] = rowmax[(g << 3) + i] + rowlse[(g << 3) + i];
#pragma unroll 2
  for (int tile = wave; tile < 626; tile += 8) {
    int n0 = tile << 4;
    v8f acc = {};
#pragma unroll
    for (int kk = 0; kk < 4; ++kk)
      acc = wmma16(Af[kk], load_B_glb(Wout16, 128, n0, kk << 5, lane), acc);
    int n = n0 + n_lane;
    if (n < 10001) {
      float bias = bpad[n];
#pragma unroll
      for (int i = 0; i < 8; ++i) {
        size_t b = (size_t)(bb + (g << 3) + i);
        out[(b * 64 + t) * 10001 + n] = acc[i] + bias - myoff[i];
      }
    }
  }
}

extern "C" void kernel_launch(void* const* d_in, const int* in_sizes, int n_in,
                              void* d_out, int out_size, void* d_ws,
                              size_t ws_size, hipStream_t stream) {
  (void)in_sizes; (void)n_in; (void)out_size; (void)ws_size;
  const int*   inst   = (const int*)d_in[0];
  const float* prevc  = (const float*)d_in[1];
  const float* finalc = (const float*)d_in[2];
  const float* tobj   = (const float*)d_in[3];
  const float* embedW = (const float*)d_in[4];
  const float* Wih    = (const float*)d_in[5];
  const float* Whh    = (const float*)d_in[6];
  const float* bih    = (const float*)d_in[7];
  const float* bhh    = (const float*)d_in[8];
  const float* Wout   = (const float*)d_in[9];
  const float* bout   = (const float*)d_in[10];
  const float* Wtgt   = (const float*)d_in[11];
  const float* btgt   = (const float*)d_in[12];
  const float* Wobj   = (const float*)d_in[13];
  const float* bobj   = (const float*)d_in[14];
  const float* ceWih  = (const float*)d_in[15];
  const float* ceWhh  = (const float*)d_in[16];
  const float* cebih  = (const float*)d_in[17];
  const float* cebhh  = (const float*)d_in[18];

  char* ws = (char*)d_ws;
  size_t off = 0;
  auto alloc = [&](size_t bytes) -> void* {
    void* p = ws + off;
    off = (off + bytes + 255) & ~(size_t)255;
    return p;
  };
  _Float16*      Wout16 = (_Float16*)alloc((size_t)10016 * 128 * 2);
  unsigned char* Wout8  = (unsigned char*)alloc((size_t)10016 * 128);
  float*         bpad   = (float*)alloc((size_t)10016 * 4);
  _Float16*      Wmain  = (_Float16*)alloc((size_t)512 * 192 * 2);
  float*         bmain  = (float*)alloc((size_t)512 * 4);
  _Float16*      Wce    = (_Float16*)alloc((size_t)256 * 128 * 2);
  float*         bce    = (float*)alloc((size_t)256 * 4);
  float*         enc    = (float*)alloc((size_t)256 * 64 * 4);
  _Float16*      Hseq   = (_Float16*)alloc((size_t)64 * 256 * 128 * 2);

  prep_wout_kernel<<<(10016 * 128 + 255) / 256, 256, 0, stream>>>(
      Wout, bout, Wout16, Wout8, bpad);
  prep_small_kernel<<<(512 * 192 + 255) / 256, 256, 0, stream>>>(
      Wih, Whh, bih, bhh, ceWih, ceWhh, cebih, cebhh, Wmain, bmain, Wce, bce);
  canvas_encode_kernel<<<16, 256, 0, stream>>>(prevc, finalc, Wobj, bobj, Wce,
                                               bce, enc);
  main_lstm_kernel<<<16, 256, 0, stream>>>(inst, tobj, Wtgt, btgt, embedW, enc,
                                           Wmain, bmain, Hseq);
  logits_kernel<<<1024, 256, 0, stream>>>(Hseq, Wout16, Wout8, bpad,
                                          (float*)d_out);
}
